// MinimalAttention_29403346108603
// MI455X (gfx1250) — compile-verified, matrix operand order checked by reading, not run on hardware
//
#include <hip/hip_runtime.h>

// MinimalAttention on MI455X (gfx1250), bf16 WMMA path with f32 accumulation.
// Stage 1: qkv = x @ w_qkv + b_qkv  -> K,V fp32 to d_out; Q,K,V bf16 to ws
// Stage 2: causal flash attention (online softmax) -> attn bf16 [B,S,D] to ws
//          Q/K tiles streamed into LDS by the Tensor Data Mover (TENSORcnt)
// Stage 3: out = attn @ w_out + b_out -> fp32 to d_out
// Workspace usage: 4 * 16777216 bf16 = 128 MiB.

#define DM    2048
#define NQKV  6144
#define NH    16
#define DH    128
#define BB    4
#define SS    2048
#define ROWS  (BB * SS)           // 8192
#define NELEM (BB * NH * SS * DH) // 16777216

typedef __bf16 bf16;
typedef __bf16 bf16x16 __attribute__((ext_vector_type(16)));
typedef __bf16 bf16x8  __attribute__((ext_vector_type(8)));
typedef float  f32x8   __attribute__((ext_vector_type(8)));

#if defined(__has_builtin)
#if __has_builtin(__builtin_amdgcn_tensor_load_to_lds)
#define HAVE_TDM 1
#endif
#endif

#ifdef HAVE_TDM
typedef unsigned int v4u __attribute__((ext_vector_type(4)));
typedef int          v8i __attribute__((ext_vector_type(8)));
typedef int          v4i __attribute__((ext_vector_type(4)));

// TDM 1-D copy of `n` bf16 elements (n <= 65535) from global to LDS.
// D# per ISA 8.3/8.4: group0 = {count=1, lds_addr, global_addr[56:0], type=2},
// group1 = {data_size=1(2B), tensor_dim0=n, tensor_dim1=1, tile_dim0=n,
//           tile_dim1=1, dim0_stride=n}. Issued once per wave; TENSORcnt.
// Toolchain here is the 6-arg clang-23 form:
//   (uint32x4 g0, int32x8 g1, int32x4 g2, int32x4 g3, int32x8 pad, i32 cpol)
__device__ __forceinline__ void tdm_copy_bf16(void* lds_ptr, const void* gptr,
                                              unsigned n) {
  unsigned long long ga = (unsigned long long)gptr;
  v4u g0;
  g0.x = 1u;                                   // count=1, no gather
  g0.y = (unsigned)(unsigned long long)lds_ptr;  // lds_addr (bytes)
  g0.z = (unsigned)ga;                         // global_addr[31:0]
  g0.w = (unsigned)(ga >> 32) | 0x80000000u;   // addr[56:32] | type=2<<30
  v8i g1;
  g1[0] = (int)(1u << 16);                     // wg_mask=0, data_size=1 (2B)
  g1[1] = (int)((n & 0xFFFFu) << 16);          // tensor_dim0 lo16 @bit48
  g1[2] = (int)(((n >> 16) & 0xFFFFu) | (1u << 16)); // dim0 hi, tensor_dim1=1
  g1[3] = (int)((n & 0xFFFFu) << 16);          // tile_dim0 @bit112
  g1[4] = 1;                                   // tile_dim1=1, tile_dim2=0
  g1[5] = (int)n;                              // tensor_dim0_stride lo32
  g1[6] = 0;
  g1[7] = 0;
  v4i z4 = {0, 0, 0, 0};
  v8i z8 = {0, 0, 0, 0, 0, 0, 0, 0};
  __builtin_amdgcn_tensor_load_to_lds(g0, g1, z4, z4, z8, 0);
}
#endif

// Load a 16x32 bf16 fragment (A-matrix layout, ISA 7.12.2) from an LDS tile
// stored row-major with `stride` bf16 per row. Same pattern serves B when the
// tile is stored [n][k].
__device__ __forceinline__ bf16x16 load_frag(const bf16* base, int stride) {
  const int lane = threadIdx.x & 31;
  const int half = lane >> 4;
  const int m    = lane & 15;
  const bf16* p0 = base + m * stride + half * 8;
  bf16x8 lo = *(const bf16x8*)p0;
  bf16x8 hi = *(const bf16x8*)(p0 + 16);
  bf16x16 r;
#pragma unroll
  for (int i = 0; i < 8; ++i) { r[i] = lo[i]; r[i + 8] = hi[i]; }
  return r;
}

__device__ __forceinline__ f32x8 wmma_bf16(bf16x16 a, bf16x16 b, f32x8 c) {
  return __builtin_amdgcn_wmma_f32_16x16x32_bf16(false, a, false, b, (short)0,
                                                 c, false, false);
}

// ---------------------------------------------------------------------------
// Stage 1: QKV projection. Block = 128 threads (4 waves), 128x64 output tile.
// Wave w owns rows w*32..w*32+31 x all 64 cols: 2x4 WMMA accumulators.
// ---------------------------------------------------------------------------
__global__ __launch_bounds__(128) void qkv_gemm(
    const float* __restrict__ x, const float* __restrict__ w,
    const float* __restrict__ bias, float* __restrict__ k_out,
    float* __restrict__ v_out, bf16* __restrict__ q_ws,
    bf16* __restrict__ k_ws, bf16* __restrict__ v_ws) {
  __shared__ alignas(16) bf16 As[128][32];  // x tile, row-major [m][k]
  __shared__ alignas(16) bf16 Bs[64][32];   // w tile, transposed [n][k]
  const int tid = threadIdx.x;
  const int wave = tid >> 5, lane = tid & 31;
  const int row0 = blockIdx.y * 128;
  const int col0 = blockIdx.x * 64;
  f32x8 acc[2][4] = {};

  for (int kt = 0; kt < DM / 32; ++kt) {
    // A: 128x32 fp32, coalesced b128 loads, cvt to bf16
#pragma unroll
    for (int i = 0; i < 8; ++i) {
      int g = i * 128 + tid;
      int r = g >> 3, c4 = (g & 7) * 4;
      const float4 v = *(const float4*)&x[(size_t)(row0 + r) * DM + kt * 32 + c4];
      As[r][c4 + 0] = (bf16)v.x; As[r][c4 + 1] = (bf16)v.y;
      As[r][c4 + 2] = (bf16)v.z; As[r][c4 + 3] = (bf16)v.w;
    }
    // B: 32x64 fp32 coalesced, stored transposed into LDS as [n][k]
#pragma unroll
    for (int i = 0; i < 4; ++i) {
      int g = i * 128 + tid;
      int kk = g >> 4, c4 = (g & 15) * 4;
      const float4 v =
          *(const float4*)&w[(size_t)(kt * 32 + kk) * NQKV + col0 + c4];
      Bs[c4 + 0][kk] = (bf16)v.x; Bs[c4 + 1][kk] = (bf16)v.y;
      Bs[c4 + 2][kk] = (bf16)v.z; Bs[c4 + 3][kk] = (bf16)v.w;
    }
    // Prefetch next K-step tiles into cache while WMMAs run
    if (kt + 1 < DM / 32) {
      __builtin_prefetch(
          &x[(size_t)(row0 + (tid >> 3)) * DM + (kt + 1) * 32 + (tid & 7) * 4],
          0, 0);
      __builtin_prefetch(
          &w[(size_t)((kt + 1) * 32 + (tid >> 4)) * NQKV + col0 + (tid & 15) * 4],
          0, 0);
    }
    __syncthreads();
    bf16x16 a0 = load_frag(&As[wave * 32][0], 32);
    bf16x16 a1 = load_frag(&As[wave * 32 + 16][0], 32);
#pragma unroll
    for (int j = 0; j < 4; ++j) {
      bf16x16 b = load_frag(&Bs[j * 16][0], 32);
      acc[0][j] = wmma_bf16(a0, b, acc[0][j]);
      acc[1][j] = wmma_bf16(a1, b, acc[1][j]);
    }
    __syncthreads();
  }

  // Epilogue: C layout lane=(n + 16*(m/8)), vgpr r -> m = r + 8*(lane>=16)
  const int nlane = lane & 15, half = lane >> 4;
#pragma unroll
  for (int j = 0; j < 4; ++j) {
    int gcol = col0 + j * 16 + nlane;  // 0..6143
    float bv = bias[gcol];
    int sec = gcol >> 11;              // 0=Q, 1=K, 2=V
    int within = gcol & 2047;
    int h = within >> 7, dh = within & 127;
#pragma unroll
    for (int i = 0; i < 2; ++i) {
#pragma unroll
      for (int r = 0; r < 8; ++r) {
        int grow = row0 + wave * 32 + i * 16 + r + half * 8;
        float val = acc[i][j][r] + bv;
        int b = grow >> 11, s = grow & 2047;
        size_t idx = ((size_t)(b * NH + h) * SS + s) * DH + dh;
        if (sec == 0) {
          q_ws[idx] = (bf16)val;
        } else if (sec == 1) {
          k_out[idx] = val;  k_ws[idx] = (bf16)val;
        } else {
          v_out[idx] = val;  v_ws[idx] = (bf16)val;
        }
      }
    }
  }
}

// ---------------------------------------------------------------------------
// Stage 2: causal flash attention. Block = 128 threads (4 waves).
// blockIdx.x = query tile (64 rows), blockIdx.y = b*NH + h.
// Q/K tiles are contiguous bf16 blocks -> TDM async copy to LDS (wave 0),
// V tile is transposed through VGPRs (overlaps the DMA).
// ---------------------------------------------------------------------------
__global__ __launch_bounds__(128) void flash_attn(
    const bf16* __restrict__ q_ws, const bf16* __restrict__ k_ws,
    const bf16* __restrict__ v_ws, bf16* __restrict__ attn_ws) {
  __shared__ alignas(16) bf16 Qs[64][128];   // [q][dh]      16 KB
  __shared__ alignas(16) bf16 Ks[32][128];   // [key][dh]     8 KB
  __shared__ alignas(16) bf16 Vts[128][32];  // [dh][key]     8 KB
  __shared__ alignas(16) bf16 Ps[4][16][32]; // per-wave P    4 KB
  const int tid = threadIdx.x, wave = tid >> 5, lane = tid & 31;
  const int nlane = lane & 15, half = lane >> 4;
  const int bh = blockIdx.y;
  const int q0 = blockIdx.x * 64;
  const size_t base = (size_t)bh * SS * DH;

  // Q tile: 64x128 bf16 (contiguous 16 KB)
#ifdef HAVE_TDM
  if (tid < 32) {  // wave 0 drives the DMA; TDM ignores EXEC
    tdm_copy_bf16(&Qs[0][0], &q_ws[base + (size_t)q0 * DH], 64 * DH);
  }
#else
#pragma unroll
  for (int i = 0; i < 8; ++i) {
    int g = i * 128 + tid;
    int r = g >> 4, c8 = (g & 15) * 8;
    *(bf16x8*)&Qs[r][c8] =
        *(const bf16x8*)&q_ws[base + (size_t)(q0 + r) * DH + c8];
  }
#endif

  f32x8 O[8] = {};
  float m_run[8], l_run[8];
#pragma unroll
  for (int r = 0; r < 8; ++r) { m_run[r] = -__builtin_inff(); l_run[r] = 0.f; }
  const float sc = 0.08838834764831845f;  // 1/sqrt(128)
  const int qrow_base = q0 + wave * 16 + half * 8;

  const int ktiles = (q0 + 95) / 32;  // cover keys 0..q0+63
  for (int kt = 0; kt < ktiles; ++kt) {
    const int k0 = kt * 32;
    // K tile 32x128 bf16 (contiguous 8 KB) -> DMA to LDS
#ifdef HAVE_TDM
    if (tid < 32) {
      tdm_copy_bf16(&Ks[0][0], &k_ws[base + (size_t)k0 * DH], 32 * DH);
    }
#else
#pragma unroll
    for (int i = 0; i < 4; ++i) {
      int g = i * 128 + tid;
      int r = g >> 4, c8 = (g & 15) * 8;
      *(bf16x8*)&Ks[r][c8] =
          *(const bf16x8*)&k_ws[base + (size_t)(k0 + r) * DH + c8];
    }
#endif
    // V tile transposed into [dh][key] through VGPRs (overlaps the DMA)
#pragma unroll
    for (int i = 0; i < 4; ++i) {
      int g = i * 128 + tid;
      int r = g >> 4, c8 = (g & 15) * 8;
      bf16x8 v = *(const bf16x8*)&v_ws[base + (size_t)(k0 + r) * DH + c8];
#pragma unroll
      for (int j = 0; j < 8; ++j) Vts[c8 + j][r] = v[j];
    }
#ifdef HAVE_TDM
    if (tid < 32) __builtin_amdgcn_s_wait_tensorcnt(0);
#endif
    __syncthreads();

    // Scores: 16 q-rows x 32 keys per wave, K-dim = dh in 4 steps of 32
    f32x8 s0 = {}, s1 = {};
#pragma unroll
    for (int d = 0; d < 4; ++d) {
      bf16x16 aq  = load_frag(&Qs[wave * 16][d * 32], 128);
      bf16x16 bk0 = load_frag(&Ks[0][d * 32], 128);
      bf16x16 bk1 = load_frag(&Ks[16][d * 32], 128);
      s0 = wmma_bf16(aq, bk0, s0);
      s1 = wmma_bf16(aq, bk1, s1);
    }

    const int kg0 = k0 + nlane, kg1 = k0 + 16 + nlane;
#pragma unroll
    for (int r = 0; r < 8; ++r) {
      const int qg = qrow_base + r;
      float v0 = (kg0 <= qg) ? s0[r] * sc : -3.0e38f;
      float v1 = (kg1 <= qg) ? s1[r] * sc : -3.0e38f;
      float mx = fmaxf(v0, v1);
#pragma unroll
      for (int msk = 8; msk >= 1; msk >>= 1)
        mx = fmaxf(mx, __shfl_xor(mx, msk, 32));
      float mnew = fmaxf(m_run[r], mx);
      float corr = expf(m_run[r] - mnew);
      float p0 = expf(v0 - mnew), p1 = expf(v1 - mnew);
      float rs = p0 + p1;
#pragma unroll
      for (int msk = 8; msk >= 1; msk >>= 1) rs += __shfl_xor(rs, msk, 32);
      l_run[r] = l_run[r] * corr + rs;
      m_run[r] = mnew;
#pragma unroll
      for (int t = 0; t < 8; ++t) O[t][r] *= corr;
      // stage P for C-layout -> A-layout conversion (per-wave LDS, DS in-order)
      Ps[wave][r + half * 8][nlane]      = (bf16)p0;
      Ps[wave][r + half * 8][nlane + 16] = (bf16)p1;
    }

    // O += P (16x32) @ V (32x128)
    bf16x16 ap = load_frag(&Ps[wave][0][0], 32);
#pragma unroll
    for (int t = 0; t < 8; ++t) {
      bf16x16 bv = load_frag(&Vts[t * 16][0], 32);
      O[t] = wmma_bf16(ap, bv, O[t]);
    }
    __syncthreads();
  }

  // Normalize and store to attn ws as [B,S,D] bf16
  const int b = bh >> 4, h = bh & 15;
#pragma unroll
  for (int r = 0; r < 8; ++r) {
    float inv = 1.0f / l_run[r];
    int qg = qrow_base + r;
    size_t rowoff = ((size_t)b * SS + qg) * DM + h * DH;
#pragma unroll
    for (int t = 0; t < 8; ++t)
      attn_ws[rowoff + t * 16 + nlane] = (bf16)(O[t][r] * inv);
  }
}

// ---------------------------------------------------------------------------
// Stage 3: output projection (attn bf16 @ w_out fp32 + b_out -> fp32).
// Same 128x64 tiling as stage 1.
// ---------------------------------------------------------------------------
__global__ __launch_bounds__(128) void out_gemm(
    const bf16* __restrict__ attn, const float* __restrict__ w,
    const float* __restrict__ bias, float* __restrict__ out) {
  __shared__ alignas(16) bf16 As[128][32];
  __shared__ alignas(16) bf16 Bs[64][32];
  const int tid = threadIdx.x, wave = tid >> 5, lane = tid & 31;
  const int row0 = blockIdx.y * 128, col0 = blockIdx.x * 64;
  f32x8 acc[2][4] = {};

  for (int kt = 0; kt < DM / 32; ++kt) {
    // A: 128x32 bf16, 16-byte loads
#pragma unroll
    for (int i = 0; i < 4; ++i) {
      int g = i * 128 + tid;
      int r = g >> 2, c8 = (g & 3) * 8;
      *(bf16x8*)&As[r][c8] =
          *(const bf16x8*)&attn[(size_t)(row0 + r) * DM + kt * 32 + c8];
    }
#pragma unroll
    for (int i = 0; i < 4; ++i) {
      int g = i * 128 + tid;
      int kk = g >> 4, c4 = (g & 15) * 4;
      const float4 v =
          *(const float4*)&w[(size_t)(kt * 32 + kk) * DM + col0 + c4];
      Bs[c4 + 0][kk] = (bf16)v.x; Bs[c4 + 1][kk] = (bf16)v.y;
      Bs[c4 + 2][kk] = (bf16)v.z; Bs[c4 + 3][kk] = (bf16)v.w;
    }
    if (kt + 1 < DM / 32) {
      __builtin_prefetch(
          &attn[(size_t)(row0 + (tid >> 2)) * DM + (kt + 1) * 32 + (tid & 3) * 8],
          0, 0);
      __builtin_prefetch(
          &w[(size_t)((kt + 1) * 32 + (tid >> 4)) * DM + col0 + (tid & 15) * 4],
          0, 0);
    }
    __syncthreads();
    bf16x16 a0 = load_frag(&As[wave * 32][0], 32);
    bf16x16 a1 = load_frag(&As[wave * 32 + 16][0], 32);
#pragma unroll
    for (int j = 0; j < 4; ++j) {
      bf16x16 b = load_frag(&Bs[j * 16][0], 32);
      acc[0][j] = wmma_bf16(a0, b, acc[0][j]);
      acc[1][j] = wmma_bf16(a1, b, acc[1][j]);
    }
    __syncthreads();
  }

  const int nlane = lane & 15, half = lane >> 4;
#pragma unroll
  for (int j = 0; j < 4; ++j) {
    int gcol = col0 + j * 16 + nlane;
    float bv = bias[gcol];
#pragma unroll
    for (int i = 0; i < 2; ++i) {
#pragma unroll
      for (int r = 0; r < 8; ++r) {
        int grow = row0 + wave * 32 + i * 16 + r + half * 8;
        out[(size_t)grow * DM + gcol] = acc[i][j][r] + bv;
      }
    }
  }
}

extern "C" void kernel_launch(void* const* d_in, const int* in_sizes, int n_in,
                              void* d_out, int out_size, void* d_ws,
                              size_t ws_size, hipStream_t stream) {
  const float* x     = (const float*)d_in[0];
  const float* w_qkv = (const float*)d_in[1];
  const float* b_qkv = (const float*)d_in[2];
  const float* w_out = (const float*)d_in[3];
  const float* b_out = (const float*)d_in[4];

  float* out   = (float*)d_out;
  float* k_out = out + (size_t)NELEM;
  float* v_out = out + 2 * (size_t)NELEM;

  bf16* q_ws    = (bf16*)d_ws;
  bf16* k_ws    = q_ws + (size_t)NELEM;
  bf16* v_ws    = k_ws + (size_t)NELEM;
  bf16* attn_ws = v_ws + (size_t)NELEM;  // [B,S,D] bf16

  qkv_gemm<<<dim3(NQKV / 64, ROWS / 128), 128, 0, stream>>>(
      x, w_qkv, b_qkv, k_out, v_out, q_ws, k_ws, v_ws);
  flash_attn<<<dim3(SS / 64, BB * NH), 128, 0, stream>>>(q_ws, k_ws, v_ws,
                                                         attn_ws);
  out_gemm<<<dim3(DM / 64, ROWS / 128), 128, 0, stream>>>(attn_ws, w_out, b_out,
                                                          out);
}
